// AggregateConcatenate_81406810129026
// MI455X (gfx1250) — matrix operand order repaired and checked by reading, not compiled, not run on hardware
//
#include <hip/hip_runtime.h>
#include <hip/hip_bf16.h>

typedef _Float16 v16h __attribute__((ext_vector_type(16)));
typedef _Float16 v8h  __attribute__((ext_vector_type(8)));
typedef _Float16 h4   __attribute__((ext_vector_type(4)));
typedef float    v8f  __attribute__((ext_vector_type(8)));

#define E_DIM 512
#define H_DIM 1024
#define A_DIM 512
#define T_DIM 2048
#define B_DIM 32
#define OUT_T 2052   // 4 + T

// ---- order-preserving float<->uint key for integer atomic max/min ----
__device__ __forceinline__ unsigned fkey(float f) {
    unsigned u = __float_as_uint(f);
    return (u & 0x80000000u) ? ~u : (u | 0x80000000u);
}
__device__ __forceinline__ float kinv(unsigned k) {
    unsigned u = (k & 0x80000000u) ? (k & 0x7fffffffu) : ~k;
    return __uint_as_float(u);
}

__device__ __forceinline__ v16h make16(v8h lo, v8h hi) {
    return __builtin_shufflevector(lo, hi, 0,1,2,3,4,5,6,7,8,9,10,11,12,13,14,15);
}

// ---- fp32 -> f16 weight conversion (all four matrices are 524288 elems) ----
__global__ __launch_bounds__(256) void convert_weights(
    const float* __restrict__ a, const float* __restrict__ b,
    const float* __restrict__ c, const float* __restrict__ d,
    _Float16* __restrict__ oa, _Float16* __restrict__ ob,
    _Float16* __restrict__ oc, _Float16* __restrict__ od)
{
    int i = blockIdx.x * 256 + threadIdx.x;           // over float4s: 524288/4
    if (i >= 524288 / 4) return;
    float4 fa = ((const float4*)a)[i];
    float4 fb = ((const float4*)b)[i];
    float4 fc = ((const float4*)c)[i];
    float4 fd = ((const float4*)d)[i];
    h4 ha, hb, hc, hd;
    ha[0]=(_Float16)fa.x; ha[1]=(_Float16)fa.y; ha[2]=(_Float16)fa.z; ha[3]=(_Float16)fa.w;
    hb[0]=(_Float16)fb.x; hb[1]=(_Float16)fb.y; hb[2]=(_Float16)fb.z; hb[3]=(_Float16)fb.w;
    hc[0]=(_Float16)fc.x; hc[1]=(_Float16)fc.y; hc[2]=(_Float16)fc.z; hc[3]=(_Float16)fc.w;
    hd[0]=(_Float16)fd.x; hd[1]=(_Float16)fd.y; hd[2]=(_Float16)fd.z; hd[3]=(_Float16)fd.w;
    ((h4*)oa)[i] = ha; ((h4*)ob)[i] = hb; ((h4*)oc)[i] = hc; ((h4*)od)[i] = hd;
}

// ---- zero/seed the aggregation accumulators every call ----
__global__ __launch_bounds__(256) void init_stats(
    float* __restrict__ gsum, float* __restrict__ gsq,
    unsigned* __restrict__ gmax, unsigned* __restrict__ gmin)
{
    int i = blockIdx.x * 256 + threadIdx.x;
    if (i >= B_DIM * A_DIM) return;
    gsum[i] = 0.f;
    gsq[i]  = 0.f;
    gmax[i] = 0x007FFFFFu;   // fkey(-inf)
    gmin[i] = 0xFF800000u;   // fkey(+inf)
}

// ---- fused 2-layer MLP: x[32rows,512] -> ELU -> [32,1024] -> tanh -> [32,512]
// AGG=true : reduce masked sum/sumsq/max/min per (b, a) via atomics
// AGG=false: write tanh result to out[b, 4+t, a]
template<bool AGG>
__global__ __launch_bounds__(256, 1) void mlp_fused(
    const float* __restrict__ x, const int* __restrict__ lengths,
    const _Float16* __restrict__ W1h, const float* __restrict__ b1,
    const _Float16* __restrict__ W2h, const float* __restrict__ b2,
    float* __restrict__ out,
    float* __restrict__ gsum, float* __restrict__ gsq,
    unsigned* __restrict__ gmaxk, unsigned* __restrict__ gmink)
{
    __shared__ _Float16 xs[32][520];   // x tile, f16, padded row stride
    __shared__ _Float16 hs[32][136];   // current 128-wide hidden chunk, f16

    const int tid  = threadIdx.x;
    const int wave = tid >> 5;
    const int lane = tid & 31;
    const int half = lane >> 4;
    const int l16  = lane & 15;
    const int m0   = blockIdx.x << 5;        // first global row of this tile
    const int b    = m0 >> 11;               // / T
    const int t0   = m0 & (T_DIM - 1);

    // stage x tile fp32 -> f16 into LDS (32 x 512)
    #pragma unroll
    for (int it = 0; it < 16; ++it) {
        int idx = tid + (it << 8);           // 0..4095 float4 slots
        int r = idx >> 7, c4 = idx & 127;
        float4 f = ((const float4*)x)[((size_t)(m0 + r) << 7) + c4];
        h4 o;
        o[0]=(_Float16)f.x; o[1]=(_Float16)f.y; o[2]=(_Float16)f.z; o[3]=(_Float16)f.w;
        *(h4*)&xs[r][c4 << 2] = o;
    }

    // layer-2 accumulators: wave owns N-tiles {wave, wave+8, wave+16, wave+24}
    v8f acc[4][2];
    #pragma unroll
    for (int i = 0; i < 4; ++i)
        #pragma unroll
        for (int mt = 0; mt < 2; ++mt)
            #pragma unroll
            for (int e = 0; e < 8; ++e) acc[i][mt][e] = 0.f;

    __syncthreads();

    #pragma unroll 1
    for (int ch = 0; ch < 8; ++ch) {         // H processed in chunks of 128
        const int hbase = ch << 7;

        // ---- layer 1: this wave computes 16 hidden cols x 32 rows ----
        v8f c1[2];
        #pragma unroll
        for (int mt = 0; mt < 2; ++mt)
            #pragma unroll
            for (int e = 0; e < 8; ++e) c1[mt][e] = 0.f;

        const _Float16* wrow1 = W1h + ((size_t)(hbase + (wave << 4) + l16) << 9); // *E
        #pragma unroll
        for (int kc = 0; kc < 16; ++kc) {    // E/32 k-steps
            const int kb = kc << 5;
            v16h bm = *(const v16h*)(wrow1 + kb + (half << 4));
            #pragma unroll
            for (int mt = 0; mt < 2; ++mt) {
                const int m = (mt << 4) + l16;
                v8h alo = *(const v8h*)&xs[m][kb + (half << 4)];
                v8h ahi = *(const v8h*)&xs[m][kb + 8 + (half << 4)];
                c1[mt] = __builtin_amdgcn_wmma_f32_16x16x32_f16(
                    false, make16(alo, ahi), false, bm, (short)0, c1[mt], false, false);
            }
        }

        __syncthreads();                     // prev chunk's layer-2 reads done
        {
            const int col = (wave << 4) + l16;
            const float bb = b1[hbase + col];
            #pragma unroll
            for (int mt = 0; mt < 2; ++mt)
                #pragma unroll
                for (int r = 0; r < 8; ++r) {
                    float v = c1[mt][r] + bb;
                    v = v > 0.f ? v : expm1f(v);    // ELU(alpha=1)
                    hs[(mt << 4) + (half << 3) + r][col] = (_Float16)v;
                }
        }
        __syncthreads();

        // ---- layer 2 partial: consume K=128 slice from hs ----
        #pragma unroll
        for (int i = 0; i < 4; ++i) {
            const int nt = wave + (i << 3);
            const _Float16* wrow2 = W2h + ((size_t)((nt << 4) + l16) << 10) + hbase; // *H
            #pragma unroll
            for (int kc = 0; kc < 4; ++kc) {
                const int kb = kc << 5;
                v16h bm = *(const v16h*)(wrow2 + kb + (half << 4));
                #pragma unroll
                for (int mt = 0; mt < 2; ++mt) {
                    const int m = (mt << 4) + l16;
                    v8h alo = *(const v8h*)&hs[m][kb + (half << 4)];
                    v8h ahi = *(const v8h*)&hs[m][kb + 8 + (half << 4)];
                    acc[i][mt] = __builtin_amdgcn_wmma_f32_16x16x32_f16(
                        false, make16(alo, ahi), false, bm, (short)0, acc[i][mt], false, false);
                }
            }
        }
    }

    // ---- epilogue ----
    int len = 0;
    if (AGG) len = lengths[b];
    #pragma unroll
    for (int i = 0; i < 4; ++i) {
        const int col = ((wave + (i << 3)) << 4) + l16;
        const float bb = b2[col];
        if (!AGG) {
            #pragma unroll
            for (int mt = 0; mt < 2; ++mt) {
                const int trow = t0 + (mt << 4) + (half << 3);
                float* orow = out + ((size_t)b * OUT_T + 4 + trow) * A_DIM + col;
                #pragma unroll
                for (int r = 0; r < 8; ++r)
                    orow[(size_t)r * A_DIM] = tanhf(acc[i][mt][r] + bb);
            }
        } else {
            float s = 0.f, sq = 0.f, mx = -__builtin_inff(), mn = __builtin_inff();
            #pragma unroll
            for (int mt = 0; mt < 2; ++mt)
                #pragma unroll
                for (int r = 0; r < 8; ++r) {
                    const int t = t0 + (mt << 4) + (half << 3) + r;
                    float v = tanhf(acc[i][mt][r] + bb);
                    if (t < len) { s += v; sq += v * v; mx = fmaxf(mx, v); mn = fminf(mn, v); }
                }
            // fold the two lane-halves (same column) — wave32 shuffle
            s  += __shfl_down(s, 16, 32);
            sq += __shfl_down(sq, 16, 32);
            mx  = fmaxf(mx, __shfl_down(mx, 16, 32));
            mn  = fminf(mn, __shfl_down(mn, 16, 32));
            if (half == 0) {
                const int g = (b << 9) + col;
                atomicAdd(&gsum[g], s);
                atomicAdd(&gsq[g], sq);
                atomicMax(&gmaxk[g], fkey(mx));
                atomicMin(&gmink[g], fkey(mn));
            }
        }
    }
}

// ---- rows 0..3 of the output: mean / max / min / std ----
__global__ __launch_bounds__(256) void finalize_stats(
    const int* __restrict__ lengths,
    const float* __restrict__ gsum, const float* __restrict__ gsq,
    const unsigned* __restrict__ gmaxk, const unsigned* __restrict__ gmink,
    float* __restrict__ out)
{
    int i = blockIdx.x * 256 + threadIdx.x;
    if (i >= B_DIM * A_DIM) return;
    int b = i >> 9, a = i & 511;
    float n    = (float)lengths[b];
    float s    = gsum[i];
    float mean = s / n;
    float var  = (gsq[i] - n * mean * mean) / (n - 1.0f);
    float sd   = sqrtf(fmaxf(var, 0.f));
    size_t base = ((size_t)b * OUT_T) * A_DIM + a;
    out[base]            = mean;
    out[base + A_DIM]    = kinv(gmaxk[i]);
    out[base + 2*A_DIM]  = kinv(gmink[i]);
    out[base + 3*A_DIM]  = sd;
}

extern "C" void kernel_launch(void* const* d_in, const int* in_sizes, int n_in,
                              void* d_out, int out_size, void* d_ws, size_t ws_size,
                              hipStream_t stream) {
    const float* x      = (const float*)d_in[0];
    const int*   lens   = (const int*)  d_in[1];
    const float* agg_W1 = (const float*)d_in[2];
    const float* agg_b1 = (const float*)d_in[3];
    const float* agg_W2 = (const float*)d_in[4];
    const float* agg_b2 = (const float*)d_in[5];
    const float* adj_W1 = (const float*)d_in[6];
    const float* adj_b1 = (const float*)d_in[7];
    const float* adj_W2 = (const float*)d_in[8];
    const float* adj_b2 = (const float*)d_in[9];
    float* out = (float*)d_out;

    char* ws = (char*)d_ws;
    float*    gsum = (float*)(ws);
    float*    gsq  = (float*)(ws + 65536);
    unsigned* gmax = (unsigned*)(ws + 131072);
    unsigned* gmin = (unsigned*)(ws + 196608);
    _Float16* w1a  = (_Float16*)(ws + 262144);
    _Float16* w2a  = w1a + 524288;
    _Float16* w1j  = w2a + 524288;
    _Float16* w2j  = w1j + 524288;

    convert_weights<<<512, 256, 0, stream>>>(agg_W1, agg_W2, adj_W1, adj_W2,
                                             w1a, w2a, w1j, w2j);
    init_stats<<<64, 256, 0, stream>>>(gsum, gsq, gmax, gmin);

    const int nblk = (B_DIM * T_DIM) / 32;   // 2048 row-tiles per MLP
    mlp_fused<true ><<<nblk, 256, 0, stream>>>(x, lens, w1a, agg_b1, w2a, agg_b2,
                                               nullptr, gsum, gsq, gmax, gmin);
    mlp_fused<false><<<nblk, 256, 0, stream>>>(x, lens, w1j, adj_b1, w2j, adj_b2,
                                               out, nullptr, nullptr, nullptr, nullptr);

    finalize_stats<<<64, 256, 0, stream>>>(lens, gsum, gsq, gmax, gmin, out);
}
